// CustomKMeansAttention_8022998909309
// MI455X (gfx1250) — matrix operand order repaired and checked by reading, not compiled, not run on hardware
//
#include <hip/hip_runtime.h>
#include <math.h>

// ---------------------------------------------------------------------------
// CDNA5 (gfx1250) wave32 WMMA implementation of k-means-sampled attention.
// Matrix math: v_wmma_f32_16x16x32_f16. Q-tile staged via TENSOR_LOAD_TO_LDS
// (TDM); f16 tiles staged via GLOBAL_LOAD_ASYNC_TO_LDS_B128 where available;
// GEMM main loops are register-pipelined to overlap global loads with WMMA.
// ---------------------------------------------------------------------------

typedef __attribute__((ext_vector_type(16))) _Float16 v16h;
typedef __attribute__((ext_vector_type(8)))  _Float16 v8h;
typedef __attribute__((ext_vector_type(4)))  _Float16 h4;
typedef __attribute__((ext_vector_type(8)))  float    v8f;
typedef __attribute__((ext_vector_type(4)))  unsigned int u32x4;
typedef __attribute__((ext_vector_type(4)))  int      i32x4;
typedef __attribute__((ext_vector_type(8)))  int      i32x8;
// The async-LDS builtins take pointers to 4 x i32 (per hipcc diagnostics).
typedef __attribute__((__vector_size__(4 * sizeof(int)))) int vsi4;

#define WMMA_F16(A, B, C) \
  __builtin_amdgcn_wmma_f32_16x16x32_f16(false, (A), false, (B), (short)0, (C), false, false)

#if defined(__has_builtin)
#  if __has_builtin(__builtin_amdgcn_tensor_load_to_lds) && \
      __has_builtin(__builtin_amdgcn_s_wait_tensorcnt)
#    define HAVE_TDM 1
#  endif
#  if __has_builtin(__builtin_amdgcn_global_load_async_to_lds_b128) && \
      __has_builtin(__builtin_amdgcn_s_wait_asynccnt)
#    define HAVE_ASYNC 1
#  endif
#endif
#ifndef HAVE_TDM
#  define HAVE_TDM 0
#endif
#ifndef HAVE_ASYNC
#  define HAVE_ASYNC 0
#endif
#if __has_include(<hip/amd_detail/amd_gfx1250_TDM.h>)
#  define TDM_6ARG 1
#else
#  define TDM_6ARG 0
#endif

static constexpr int Bb    = 4;
static constexpr int Nn    = 4096;
static constexpr int Cc    = 1024;
static constexpr int Hh    = 16;
static constexpr int Dh    = 64;
static constexpr int NS    = 64;   // sampled keys
static constexpr int NCl   = 8;    // clusters
static constexpr int ITERS = 10;
static constexpr int LDA   = 72;   // LDS leading dim (halves); 144B rows keep 16B align

__device__ inline v8f acc_zero() {
  v8f z = {0.f,0.f,0.f,0.f,0.f,0.f,0.f,0.f};
  return z;
}

// A fragment: 16x32 f16 from row-major LDS tile (leading dim `ld` halves).
// lanes 0-15 : row=lane,    halves 0-7 -> K 0-7,  8-15 -> K 16-23
// lanes 16-31: row=lane-16, halves 0-7 -> K 8-15, 8-15 -> K 24-31
__device__ inline v16h frag_a(const _Float16* lds, int row0, int k0, int ld) {
  const int lane = threadIdx.x & 31;
  const _Float16* p = lds + (row0 + (lane & 15)) * ld + k0 + ((lane >> 4) << 3);
  v8h lo = *(const v8h*)p;
  v8h hi = *(const v8h*)(p + 16);
  v16h f;
#pragma unroll
  for (int h = 0; h < 8; ++h) { f[h] = lo[h]; f[8 + h] = hi[h]; }
  return f;
}

// B fragment: 32x16 where LDS stores B^T (rows = out col, cols = K).
// lanes 0-15: col=lane, K 0-7,16-23 ; lanes 16-31: col=lane-16, K 8-15,24-31.
__device__ inline v16h frag_b(const _Float16* lds, int col0, int k0, int ld) {
  const int lane = threadIdx.x & 31;
  const _Float16* p = lds + (col0 + (lane & 15)) * ld + k0 + ((lane >> 4) << 3);
  v8h lo = *(const v8h*)p;
  v8h hi = *(const v8h*)(p + 16);
  v16h f;
#pragma unroll
  for (int h = 0; h < 8; ++h) { f[h] = lo[h]; f[8 + h] = hi[h]; }
  return f;
}

__device__ inline unsigned pcg(unsigned x) {
  x = x * 747796405u + 2891336453u;
  unsigned w = ((x >> ((x >> 28u) + 4u)) ^ x) * 277803737u;
  return (w >> 22u) ^ w;
}

// ---------------------------------------------------------------------------
// Kernel 1: qkv = x @ W_qkv^T + b ; scatter to q/k/v (B,H,N,Dh) f16 layout.
// 128x64 block tile, BK=64, 8 waves, register-pipelined stage-in.
// ---------------------------------------------------------------------------
__global__ __launch_bounds__(256)
void qkv_wmma(const float* __restrict__ X, const float* __restrict__ W,
              const float* __restrict__ bias,
              _Float16* __restrict__ qh, _Float16* __restrict__ kh,
              _Float16* __restrict__ vh) {
  __shared__ _Float16 sA[128 * LDA];
  __shared__ _Float16 sB[64 * LDA];
  const int m0 = blockIdx.y * 128;
  const int n0 = blockIdx.x * 64;
  const int tid = threadIdx.x;
  const int lane = tid & 31, wave = tid >> 5;
  const int wm = wave & 3, wn = wave >> 2;   // 4x2 wave grid -> 128x64

  v8f acc[2][2];
#pragma unroll
  for (int i = 0; i < 2; ++i)
#pragma unroll
    for (int j = 0; j < 2; ++j) acc[i][j] = acc_zero();

  float4 ra[8], rb[4];
  auto loadA = [&](int k0) {
#pragma unroll
    for (int i = 0; i < 8; ++i) {            // A: 128x64 f32, float4 loads
      int e = tid + i * 256, r = e >> 4, c4 = e & 15;
      ra[i] = *(const float4*)&X[(size_t)(m0 + r) * Cc + k0 + c4 * 4];
    }
  };
  auto loadB = [&](int k0) {
#pragma unroll
    for (int i = 0; i < 4; ++i) {            // B^T: 64x64 f32
      int e = tid + i * 256, r = e >> 4, c4 = e & 15;
      rb[i] = *(const float4*)&W[(size_t)(n0 + r) * Cc + k0 + c4 * 4];
    }
  };

  loadA(0); loadB(0);
  for (int k0 = 0; k0 < Cc; k0 += 64) {
#pragma unroll
    for (int i = 0; i < 8; ++i) {            // regs -> LDS (f32 -> f16)
      int e = tid + i * 256, r = e >> 4, c4 = e & 15;
      h4 hv = {(_Float16)ra[i].x, (_Float16)ra[i].y, (_Float16)ra[i].z, (_Float16)ra[i].w};
      *(h4*)&sA[r * LDA + c4 * 4] = hv;
    }
#pragma unroll
    for (int i = 0; i < 4; ++i) {
      int e = tid + i * 256, r = e >> 4, c4 = e & 15;
      h4 hv = {(_Float16)rb[i].x, (_Float16)rb[i].y, (_Float16)rb[i].z, (_Float16)rb[i].w};
      *(h4*)&sB[r * LDA + c4 * 4] = hv;
    }
    __syncthreads();
    if (k0 + 64 < Cc) { loadA(k0 + 64); loadB(k0 + 64); }  // overlap with WMMA
#pragma unroll
    for (int kk = 0; kk < 2; ++kk) {
      v16h fa0 = frag_a(sA, wm * 32,      kk * 32, LDA);
      v16h fa1 = frag_a(sA, wm * 32 + 16, kk * 32, LDA);
      v16h fb0 = frag_b(sB, wn * 32,      kk * 32, LDA);
      v16h fb1 = frag_b(sB, wn * 32 + 16, kk * 32, LDA);
      acc[0][0] = WMMA_F16(fa0, fb0, acc[0][0]);
      acc[0][1] = WMMA_F16(fa0, fb1, acc[0][1]);
      acc[1][0] = WMMA_F16(fa1, fb0, acc[1][0]);
      acc[1][1] = WMMA_F16(fa1, fb1, acc[1][1]);
    }
    __syncthreads();
  }

#pragma unroll
  for (int mi = 0; mi < 2; ++mi)
#pragma unroll
    for (int ni = 0; ni < 2; ++ni) {
      int col = n0 + wn * 32 + ni * 16 + (lane & 15);
      int mb  = m0 + wm * 32 + mi * 16 + ((lane >> 4) << 3);
      float bv = bias[col];
      int which = col >> 10;       // 0=q,1=k,2=v
      int rem   = col & 1023;
      int head  = rem >> 6;
      int dh    = rem & 63;
      _Float16* dst = (which == 0) ? qh : (which == 1) ? kh : vh;
#pragma unroll
      for (int r = 0; r < 8; ++r) {
        int m = mb + r;
        int b = m >> 12, n = m & 4095;
        size_t o = (((size_t)(b * Hh + head)) * Nn + n) * Dh + dh;
        dst[o] = (_Float16)(acc[mi][ni][r] + bv);
      }
    }
}

// ---------------------------------------------------------------------------
// Kernel 2: per-(b,h) k-means + Gumbel top-64 sampling. One block per (b,h).
// ---------------------------------------------------------------------------
__global__ __launch_bounds__(256)
void kmeans_sample(const _Float16* __restrict__ kh, int* __restrict__ sidx) {
  __shared__ float cent[NCl][Dh];
  __shared__ float csum[NCl][Dh];
  __shared__ float ccnt[NCl];
  __shared__ float cn2[NCl];
  __shared__ float sc[Nn];
  __shared__ float rv[256];
  __shared__ int   ri[256];
  __shared__ float tot;
  const int bh  = blockIdx.x;
  const int tid = threadIdx.x;
  const _Float16* kp0 = kh + (size_t)bh * Nn * Dh;

  for (int e = tid; e < NCl * Dh; e += 256) {
    int c = e >> 6, d = e & 63;
    unsigned r = pcg(pcg(0x9E3779B9u ^ (unsigned)(bh * 977)) + (unsigned)c);
    int ic = (int)(r & (Nn - 1));
    cent[c][d] = (float)kp0[(size_t)ic * Dh + d];
  }
  __syncthreads();
  if (tid < NCl) {
    float s = 0.f;
    for (int d = 0; d < Dh; ++d) { float v = cent[tid][d]; s += v * v; }
    cn2[tid] = s;
  }
  __syncthreads();

  for (int it = 0; it < ITERS; ++it) {
    for (int e = tid; e < NCl * Dh; e += 256) csum[e >> 6][e & 63] = 0.f;
    if (tid < NCl) ccnt[tid] = 0.f;
    __syncthreads();
    for (int i = tid; i < Nn; i += 256) {
      const _Float16* kp = kp0 + (size_t)i * Dh;
      float kreg[Dh];
#pragma unroll
      for (int c8 = 0; c8 < Dh / 8; ++c8) {   // 128-bit key row loads
        v8h kv = *(const v8h*)(kp + c8 * 8);
#pragma unroll
        for (int j = 0; j < 8; ++j) kreg[c8 * 8 + j] = (float)kv[j];
      }
      float dot[NCl];
#pragma unroll
      for (int c = 0; c < NCl; ++c) dot[c] = 0.f;
      for (int d = 0; d < Dh; ++d) {
        float kv = kreg[d];
#pragma unroll
        for (int c = 0; c < NCl; ++c) dot[c] += kv * cent[c][d];
      }
      int a = 0; float best = cn2[0] - 2.f * dot[0];
#pragma unroll
      for (int c = 1; c < NCl; ++c) {
        float s = cn2[c] - 2.f * dot[c];
        if (s < best) { best = s; a = c; }
      }
      for (int d = 0; d < Dh; ++d) atomicAdd(&csum[a][d], kreg[d]);
      atomicAdd(&ccnt[a], 1.0f);
    }
    __syncthreads();
    for (int e = tid; e < NCl * Dh; e += 256) {
      int c = e >> 6, d = e & 63;
      float cnt = ccnt[c];
      if (cnt >= 1e-6f) cent[c][d] = csum[c][d] / (cnt + 1e-6f);
    }
    __syncthreads();
    if (tid < NCl) {
      float s = 0.f;
      for (int d = 0; d < Dh; ++d) { float v = cent[tid][d]; s += v * v; }
      cn2[tid] = s;
    }
    __syncthreads();
  }

  for (int i = tid; i < Nn; i += 256) {
    const _Float16* kp = kp0 + (size_t)i * Dh;
    float dot[NCl]; float kn = 0.f;
#pragma unroll
    for (int c = 0; c < NCl; ++c) dot[c] = 0.f;
#pragma unroll
    for (int c8 = 0; c8 < Dh / 8; ++c8) {
      v8h kvv = *(const v8h*)(kp + c8 * 8);
#pragma unroll
      for (int j = 0; j < 8; ++j) {
        float kv = (float)kvv[j]; kn += kv * kv;
#pragma unroll
        for (int c = 0; c < NCl; ++c) dot[c] += kv * cent[c][c8 * 8 + j];
      }
    }
    int a = 0; float best = cn2[0] - 2.f * dot[0];
#pragma unroll
    for (int c = 1; c < NCl; ++c) {
      float s = cn2[c] - 2.f * dot[c];
      if (s < best) { best = s; a = c; }
    }
    (void)a;
    sc[i] = sqrtf(fmaxf(kn + best, 0.f));
  }
  __syncthreads();
  { float s = 0.f; for (int i = tid; i < Nn; i += 256) s += sc[i]; rv[tid] = s; }
  __syncthreads();
  for (int s = 128; s > 0; s >>= 1) {
    if (tid < s) rv[tid] += rv[tid + s];
    __syncthreads();
  }
  if (tid == 0) tot = rv[0];
  __syncthreads();
  const float tt = tot;
  for (int i = tid; i < Nn; i += 256) {
    float p = sc[i] / (tt + 1e-6f);
    unsigned r = pcg(pcg(0x85EBCA6Bu ^ (unsigned)(bh * 131071)) + (unsigned)i);
    float u = ((float)r + 0.5f) * (1.0f / 4294967296.0f);
    float g = -logf(-logf(u));                 // Gumbel(0,1)
    sc[i] = logf(p + 1e-20f) + g;
  }
  __syncthreads();
  for (int round = 0; round < NS; ++round) {   // masked argmax top-64
    float bv = -INFINITY; int bi = 0;
    for (int i = tid; i < Nn; i += 256) {
      float v = sc[i];
      if (v > bv) { bv = v; bi = i; }
    }
    rv[tid] = bv; ri[tid] = bi;
    __syncthreads();
    for (int s = 128; s > 0; s >>= 1) {
      if (tid < s && rv[tid + s] > rv[tid]) { rv[tid] = rv[tid + s]; ri[tid] = ri[tid + s]; }
      __syncthreads();
    }
    if (tid == 0) { sidx[bh * NS + round] = ri[0]; sc[ri[0]] = -INFINITY; }
    __syncthreads();
  }
}

// ---------------------------------------------------------------------------
// Kernel 3: sampled attention. Block = (b,h) x 64-query tile, 8 waves.
// Q via TDM; K/V gather via GLOBAL_LOAD_ASYNC_TO_LDS_B128 when available.
// ---------------------------------------------------------------------------
__global__ __launch_bounds__(256)
void attn_wmma(const _Float16* __restrict__ qh, const _Float16* __restrict__ kh,
               const _Float16* __restrict__ vh, const int* __restrict__ sidx,
               _Float16* __restrict__ ah) {
  __shared__ _Float16 sK[64 * 64];    // gathered keys, s-major
  __shared__ _Float16 sVt[64 * 64];   // gathered values transposed, d-major
  __shared__ _Float16 sQ[64 * 64];
  __shared__ float    sL[64 * 64];
  __shared__ _Float16 sP[64 * 64];    // temp for V rows, then softmax probs
  __shared__ int      sI[NS];
  const int bh  = blockIdx.x;
  const int n0  = blockIdx.y * 64;
  const int tid = threadIdx.x;
  const int lane = tid & 31, wave = tid >> 5;
  const int rs = wave >> 1, ch = wave & 1;   // 4 row strips x 2 col halves
  const size_t base = (size_t)bh * Nn * Dh;

  if (tid < NS) sI[tid] = sidx[bh * NS + tid];
  __syncthreads();

  // gather K rows (s-major) and V rows (temporarily into sP)
#if HAVE_ASYNC
#pragma unroll
  for (int i = 0; i < 2; ++i) {
    int e = tid + i * 256;               // 512 chunks of 8 halves
    int s = e >> 3, c8 = e & 7;
    size_t src = base + (size_t)sI[s] * Dh + c8 * 8;
    __builtin_amdgcn_global_load_async_to_lds_b128(
        (vsi4*)&kh[src], (vsi4*)&sK[s * 64 + c8 * 8], 0, 0);
    __builtin_amdgcn_global_load_async_to_lds_b128(
        (vsi4*)&vh[src], (vsi4*)&sP[s * 64 + c8 * 8], 0, 0);
  }
#else
#pragma unroll
  for (int i = 0; i < 2; ++i) {
    int e = tid + i * 256;
    int s = e >> 3, c8 = e & 7;
    size_t src = base + (size_t)sI[s] * Dh + c8 * 8;
    *(v8h*)&sK[s * 64 + c8 * 8] = *(const v8h*)&kh[src];
    *(v8h*)&sP[s * 64 + c8 * 8] = *(const v8h*)&vh[src];
  }
#endif

#if HAVE_TDM
  // Stage Q tile (64x64 f16 rows of the (4096,64) per-(b,h) tensor) via TDM.
  if (tid < 32) {
    unsigned long long ga = (unsigned long long)(const void*)(qh + base + (size_t)n0 * Dh);
    unsigned ldsoff = (unsigned)(unsigned long long)(const void*)&sQ[0];
    u32x4 g0;
    g0[0] = 1u;                                        // count=1 user descriptor
    g0[1] = ldsoff;                                    // lds_addr
    g0[2] = (unsigned)(ga & 0xFFFFFFFFu);              // global_addr[31:0]
    g0[3] = (unsigned)((ga >> 32) & 0x1FFFFFFu) | (2u << 30);  // addr[56:32], type=2
    i32x8 g1;
    g1[0] = (int)(1u << 16);                 // workgroup_mask=0, data_size=1 (2B)
    g1[1] = (int)(64u << 16);                // tensor_dim0 = 64 (low 16 in [31:16])
    g1[2] = (int)(4096u << 16);              // tensor_dim0 hi=0 | tensor_dim1=4096 lo
    g1[3] = (int)(64u << 16);                // tensor_dim1 hi=0 | tile_dim0=64
    g1[4] = (int)64;                         // tile_dim1=64, tile_dim2=0
    g1[5] = (int)64;                         // tensor_dim0_stride = 64 (low 32)
    g1[6] = 0;                               // stride hi | tensor_dim1_stride lo
    g1[7] = 0;
    i32x4 z4 = {0, 0, 0, 0};
#if TDM_6ARG
    i32x8 z8 = {0, 0, 0, 0, 0, 0, 0, 0};
    __builtin_amdgcn_tensor_load_to_lds(g0, g1, z4, z4, z8, 0);
#else
    __builtin_amdgcn_tensor_load_to_lds(g0, g1, z4, z4, 0);
#endif
    __builtin_amdgcn_s_wait_tensorcnt(0);
  }
#else
  {
    const _Float16* qp = qh + base + (size_t)n0 * Dh;
#pragma unroll
    for (int i = 0; i < 2; ++i) {
      int e = tid + i * 256;
      *(v8h*)&sQ[e * 8] = *(const v8h*)&qp[e * 8];
    }
  }
#endif
#if HAVE_ASYNC
  __builtin_amdgcn_s_wait_asynccnt(0);
#endif
  __syncthreads();

  // transpose V rows (sP, s-major) -> sVt (d-major) in LDS
#pragma unroll
  for (int i = 0; i < 16; ++i) {
    int e = tid + i * 256;
    int s = e >> 6, d = e & 63;
    sVt[d * 64 + s] = sP[s * 64 + d];
  }
  __syncthreads();

  // logits = Q (64x64) @ K^T (64x64)
  v8f a0 = acc_zero(), a1 = acc_zero();
#pragma unroll
  for (int kk = 0; kk < 2; ++kk) {
    v16h fa  = frag_a(sQ, rs * 16, kk * 32, 64);
    v16h fb0 = frag_b(sK, ch * 32 + 0,  kk * 32, 64);
    v16h fb1 = frag_b(sK, ch * 32 + 16, kk * 32, 64);
    a0 = WMMA_F16(fa, fb0, a0);
    a1 = WMMA_F16(fa, fb1, a1);
  }
  {
    const float scale = 0.125f;   // Dh^-0.5, Dh=64
    int colb = ch * 32 + (lane & 15);
    int rowb = rs * 16 + ((lane >> 4) << 3);
#pragma unroll
    for (int r = 0; r < 8; ++r) {
      sL[(rowb + r) * 64 + colb]      = a0[r] * scale;
      sL[(rowb + r) * 64 + colb + 16] = a1[r] * scale;
    }
  }
  __syncthreads();
  if (tid < 64) {
    float mx = -INFINITY;
    for (int s = 0; s < NS; ++s) mx = fmaxf(mx, sL[tid * 64 + s]);
    float sum = 0.f;
    for (int s = 0; s < NS; ++s) {
      float e = __expf(sL[tid * 64 + s] - mx);
      sL[tid * 64 + s] = e; sum += e;
    }
    float inv = 1.0f / sum;
    for (int s = 0; s < NS; ++s) sP[tid * 64 + s] = (_Float16)(sL[tid * 64 + s] * inv);
  }
  __syncthreads();

  // out = P (64x64) @ V (64x64)
  v8f o0 = acc_zero(), o1 = acc_zero();
#pragma unroll
  for (int kk = 0; kk < 2; ++kk) {
    v16h fa  = frag_a(sP, rs * 16, kk * 32, 64);
    v16h fb0 = frag_b(sVt, ch * 32 + 0,  kk * 32, 64);
    v16h fb1 = frag_b(sVt, ch * 32 + 16, kk * 32, 64);
    o0 = WMMA_F16(fa, fb0, o0);
    o1 = WMMA_F16(fa, fb1, o1);
  }
  {
    int b = bh >> 4, h = bh & 15;
    int colb = ch * 32 + (lane & 15);
    int rowb = rs * 16 + ((lane >> 4) << 3);
#pragma unroll
    for (int r = 0; r < 8; ++r) {
      int n = n0 + rowb + r;
      size_t o = ((size_t)b * Nn + n) * Cc + h * Dh;   // (B,N,H*Dh)
      ah[o + colb]      = (_Float16)o0[r];
      ah[o + colb + 16] = (_Float16)o1[r];
    }
  }
}

// ---------------------------------------------------------------------------
// Kernel 4: out = attn_out @ W_proj^T + b_proj (f32 result), BK=64.
// A-tile (f16) staged by async LDS DMA; B register-pipelined.
// ---------------------------------------------------------------------------
__global__ __launch_bounds__(256)
void proj_wmma(const _Float16* __restrict__ A, const float* __restrict__ W,
               const float* __restrict__ bias, float* __restrict__ out) {
  __shared__ _Float16 sA[128 * LDA];
  __shared__ _Float16 sB[64 * LDA];
  const int m0 = blockIdx.y * 128;
  const int n0 = blockIdx.x * 64;
  const int tid = threadIdx.x;
  const int lane = tid & 31, wave = tid >> 5;
  const int wm = wave & 3, wn = wave >> 2;

  v8f acc[2][2];
#pragma unroll
  for (int i = 0; i < 2; ++i)
#pragma unroll
    for (int j = 0; j < 2; ++j) acc[i][j] = acc_zero();

  float4 rb[4];
  auto loadB = [&](int k0) {
#pragma unroll
    for (int i = 0; i < 4; ++i) {
      int e = tid + i * 256, r = e >> 4, c4 = e & 15;
      rb[i] = *(const float4*)&W[(size_t)(n0 + r) * Cc + k0 + c4 * 4];
    }
  };
#if HAVE_ASYNC
  auto issueA = [&](int k0) {                // async global -> LDS, 128-bit
#pragma unroll
    for (int i = 0; i < 4; ++i) {
      int e = tid + i * 256, r = e >> 3, c8 = e & 7;
      __builtin_amdgcn_global_load_async_to_lds_b128(
          (vsi4*)&A[(size_t)(m0 + r) * Cc + k0 + c8 * 8],
          (vsi4*)&sA[r * LDA + c8 * 8], 0, 0);
    }
  };
#else
  v8h raH[4];
  auto loadA = [&](int k0) {
#pragma unroll
    for (int i = 0; i < 4; ++i) {
      int e = tid + i * 256, r = e >> 3, c8 = e & 7;
      raH[i] = *(const v8h*)&A[(size_t)(m0 + r) * Cc + k0 + c8 * 8];
    }
  };
#endif

  loadB(0);
#if HAVE_ASYNC
  issueA(0);
#else
  loadA(0);
#endif
  for (int k0 = 0; k0 < Cc; k0 += 64) {
#pragma unroll
    for (int i = 0; i < 4; ++i) {            // B regs -> LDS (f32 -> f16)
      int e = tid + i * 256, r = e >> 4, c4 = e & 15;
      h4 hv = {(_Float16)rb[i].x, (_Float16)rb[i].y, (_Float16)rb[i].z, (_Float16)rb[i].w};
      *(h4*)&sB[r * LDA + c4 * 4] = hv;
    }
#if HAVE_ASYNC
    __builtin_amdgcn_s_wait_asynccnt(0);     // A tile resident in sA
#else
#pragma unroll
    for (int i = 0; i < 4; ++i) {
      int e = tid + i * 256, r = e >> 3, c8 = e & 7;
      *(v8h*)&sA[r * LDA + c8 * 8] = raH[i];
    }
#endif
    __syncthreads();
    if (k0 + 64 < Cc) {
      loadB(k0 + 64);
#if !HAVE_ASYNC
      loadA(k0 + 64);
#endif
    }
#pragma unroll
    for (int kk = 0; kk < 2; ++kk) {
      v16h fa0 = frag_a(sA, wm * 32,      kk * 32, LDA);
      v16h fa1 = frag_a(sA, wm * 32 + 16, kk * 32, LDA);
      v16h fb0 = frag_b(sB, wn * 32,      kk * 32, LDA);
      v16h fb1 = frag_b(sB, wn * 32 + 16, kk * 32, LDA);
      acc[0][0] = WMMA_F16(fa0, fb0, acc[0][0]);
      acc[0][1] = WMMA_F16(fa0, fb1, acc[0][1]);
      acc[1][0] = WMMA_F16(fa1, fb0, acc[1][0]);
      acc[1][1] = WMMA_F16(fa1, fb1, acc[1][1]);
    }
    __syncthreads();
#if HAVE_ASYNC
    if (k0 + 64 < Cc) issueA(k0 + 64);       // DMA next A tile during B stage
#endif
  }

#pragma unroll
  for (int mi = 0; mi < 2; ++mi)
#pragma unroll
    for (int ni = 0; ni < 2; ++ni) {
      int col = n0 + wn * 32 + ni * 16 + (lane & 15);
      int mb  = m0 + wm * 32 + mi * 16 + ((lane >> 4) << 3);
      float bv = bias[col];
#pragma unroll
      for (int r = 0; r < 8; ++r) {
        int m = mb + r;
        out[(size_t)m * Cc + col] = acc[mi][ni][r] + bv;
      }
    }
}

// ---------------------------------------------------------------------------
extern "C" void kernel_launch(void* const* d_in, const int* in_sizes, int n_in,
                              void* d_out, int out_size, void* d_ws, size_t ws_size,
                              hipStream_t stream) {
  const float* x      = (const float*)d_in[0];   // (B,N,C)
  const float* W_qkv  = (const float*)d_in[1];   // (3C,C)
  const float* b_qkv  = (const float*)d_in[2];   // (3C)
  const float* W_proj = (const float*)d_in[3];   // (C,C)
  const float* b_proj = (const float*)d_in[4];   // (C)
  float* out = (float*)d_out;                    // (B,N,C)
  (void)in_sizes; (void)n_in; (void)out_size; (void)ws_size;

  const size_t per = (size_t)Bb * Hh * Nn * Dh;  // 16,777,216 elems
  _Float16* qh = (_Float16*)d_ws;
  _Float16* kh = qh + per;
  _Float16* vh = kh + per;
  _Float16* ah = vh + per;                       // (B,N,C) f16 attention output
  int* sidx = (int*)(ah + per);                  // (B*H, NS)

  dim3 gq(3 * Cc / 64, (Bb * Nn) / 128);         // (48,128)
  qkv_wmma<<<gq, 256, 0, stream>>>(x, W_qkv, b_qkv, qh, kh, vh);

  kmeans_sample<<<Bb * Hh, 256, 0, stream>>>(kh, sidx);

  dim3 ga(Bb * Hh, Nn / 64);                     // (64,64)
  attn_wmma<<<ga, 256, 0, stream>>>(qh, kh, vh, sidx, ah);

  dim3 gp(Cc / 64, (Bb * Nn) / 128);             // (16,128)
  proj_wmma<<<gp, 256, 0, stream>>>(ah, W_proj, b_proj, out);
}
